// TrendGRU_44727789420876
// MI455X (gfx1250) — matrix operand order repaired and checked by reading, not compiled, not run on hardware
//
#include <hip/hip_runtime.h>

// TrendGRU on MI455X (gfx1250, wave32, WMMA).
//
// One wave owns 16 batch rows; per timestep the full (96 padded gates x 16
// batch) pre-activation is 6 x v_wmma_f32_16x16x32_f16 with the input
// projection and biases folded into extra K rows (K=24: W_ih*x, K=25: bias).
// The r/z gate tiles are additionally PRE-SCALED by 0.5 so that
// sigmoid(p) = 0.5 + 0.5*tanh(p/2) needs no pre-multiply: just v_tanh + fma.
//
// The A-tile rows are PERMUTED (tile0 rows 8-15 -> units 16-23, tile1 rows
// 0-7 -> units 8-15) so that each lane's D-layout results are exactly the
// h-values its B-layout column slice needs next step: the h^T rebuild is
// fully lane-local (no cross-lane swap on the critical path).
//
// Gate nonlinearities use the gfx1250 hardware v_tanh_f32; B is rebuilt with
// v_cvt_pk_rtz_f16_f32 + 32-bit selects.

typedef __attribute__((ext_vector_type(16))) _Float16 v16h;
typedef __attribute__((ext_vector_type(8)))  float    v8f;
typedef __attribute__((ext_vector_type(8)))  unsigned int v8u;

#define T_STEPS 512
#define HH      24

#if __has_builtin(__builtin_amdgcn_tanhf)
__device__ __forceinline__ float fast_tanh(float x) {
    return __builtin_amdgcn_tanhf(x);                    // v_tanh_f32
}
#elif __has_builtin(__builtin_amdgcn_tanh_f32)
__device__ __forceinline__ float fast_tanh(float x) {
    return __builtin_amdgcn_tanh_f32(x);                 // v_tanh_f32
}
#else
__device__ __forceinline__ float fast_tanh(float x) {
    // tanh(x) = 1 - 2/(1+exp(2x)); saturates correctly at +-1.
    return fmaf(-2.0f, __builtin_amdgcn_rcpf(1.0f + __expf(2.0f * x)), 1.0f);
}
#endif

// sigmoid of (2*d): used with A tiles pre-scaled by 0.5, so d = pre/2.
__device__ __forceinline__ float sigmoid_half_pre(float d) {
    return fmaf(0.5f, fast_tanh(d), 0.5f);               // tanh, fma
}

__device__ __forceinline__ unsigned int pk16(float lo, float hi) {
    auto p = __builtin_amdgcn_cvt_pkrtz(lo, hi);         // v_cvt_pk_rtz_f16_f32
    return __builtin_bit_cast(unsigned int, p);
}

__global__ __launch_bounds__(256) void trendgru_wmma_kernel(
    const float* __restrict__ x,     // (B, T, 1)
    const float* __restrict__ W_ih,  // (72, 1)
    const float* __restrict__ b_ih,  // (72,)
    const float* __restrict__ W_hh,  // (72, 24)
    const float* __restrict__ b_hh,  // (72,)
    const float* __restrict__ fc_w,  // (2, 24)
    const float* __restrict__ fc_b,  // (2,)
    float* __restrict__ out,         // (B, 2)
    int Bsz)
{
    const int lane    = threadIdx.x & 31;
    const int wave    = threadIdx.x >> 5;
    const int rowbase = (blockIdx.x * (blockDim.x >> 5) + wave) * 16;
    if (rowbase >= Bsz) return;                  // wave-uniform branch

    const int  col    = lane & 15;   // batch column this lane owns (N index)
    const bool hiHalf = lane >= 16;

    // ---------------------------------------------------------------
    // Constant A tiles: 6 tiles of (16 gate-rows x K=32), f16.
    // tile pair per gate: 0,1 = r ; 2,3 = z ; 4,5 = n.
    // Row->unit permutation (s = tile&1, m = row):
    //   s=0: m<8 -> unit m      ; m>=8 -> unit m+8 (16..23)
    //   s=1: m<8 -> unit m+8    ; m>=8 -> padding (zero row)
    // K = 0..23 : W_hh[unit-row, K]
    // K = 24    : W_ih[row]   (r,z only; n's x-term applied separately)
    // K = 25    : b_ih+b_hh (r,z) or b_hh (n)
    // r,z tiles (groups 0,1) are scaled by 0.5 for the tanh-based sigmoid.
    // A k order used here (contiguous, same as B): lane<16 -> K = e;
    //                                 lane>=16 -> K = e+16.
    // ---------------------------------------------------------------
    v16h A[6];
    for (int tt = 0; tt < 6; ++tt) {
        int  group = tt >> 1;                  // 0=r,1=z,2=n
        int  s     = tt & 1;
        int  m     = col;                      // this lane's A row (M index)
        int  u     = (s == 0) ? (m < 8 ? m : m + 8)
                              : (m < 8 ? m + 8 : 24 + (m - 8));
        bool valid = u < HH;
        int  grow  = group * HH + (valid ? u : 0);
        float scale = (group < 2) ? 0.5f : 1.0f;   // fold sigmoid's /2
        for (int e = 0; e < 16; ++e) {
            int K = hiHalf ? (e + 16)
                           : (e);
            float v = 0.0f;
            if (valid) {
                if      (K < HH)  v = W_hh[grow * HH + K];
                else if (K == 24) v = (group < 2) ? W_ih[grow] : 0.0f;
                else if (K == 25) v = (group < 2) ? (b_ih[grow] + b_hh[grow])
                                                  : b_hh[grow];
            }
            A[tt][e] = (_Float16)(v * scale);
        }
    }

    // n-gate x-projection constants for this lane's unit rows.
    // D tiles: lane half d, vgpr j -> tile0 unit = d ? 16+j : j
    //                                 tile1 unit = d ? pad  : 8+j
    float wn0[8], bn0[8], wn1[8], bn1[8];
    for (int j = 0; j < 8; ++j) {
        int u0 = hiHalf ? 16 + j : j;
        wn0[j] = W_ih[2 * HH + u0];
        bn0[j] = b_ih[2 * HH + u0];
        wn1[j] = hiHalf ? 0.0f : W_ih[2 * HH + 8 + j];
        bn1[j] = hiHalf ? 0.0f : b_ih[2 * HH + 8 + j];
    }

    // Hidden state, D layout with the permutation above. Padding rows have
    // zero A rows -> r=z=0.5, n=0 -> stay exactly 0 forever.
    float h0[8], h1[8];
    #pragma unroll
    for (int j = 0; j < 8; ++j) { h0[j] = 0.0f; h1[j] = 0.0f; }

    const float* xrow = x + (size_t)(rowbase + col) * T_STEPS;

    #pragma unroll 1
    for (int t = 0; t < T_STEPS; t += 4) {
        // One 16B load per 4 steps; (rowbase+col)*T is 2KB-aligned, t%4==0.
        float4 xq = *(const float4*)(xrow + t);
        float xs[4] = {xq.x, xq.y, xq.z, xq.w};

        #pragma unroll
        for (int ss = 0; ss < 4; ++ss) {
            float xv = xs[ss];

            // Build B = h^T (K=32 x N=16), f16 -- fully lane-local,
            // packed pairs + 32-bit selects.
            // lane<16 elem e -> K=e     : units 0-7 (h0), 8-15 (h1)
            // lane>=16 elem e -> K=16+e : units 16-23 (h0), then x, 1, pad
            v8u bw;
            bw[0] = pk16(h0[0], h0[1]);
            bw[1] = pk16(h0[2], h0[3]);
            bw[2] = pk16(h0[4], h0[5]);
            bw[3] = pk16(h0[6], h0[7]);
            bw[4] = hiHalf ? pk16(xv, 1.0f) : pk16(h1[0], h1[1]);
            bw[5] = hiHalf ? 0u : pk16(h1[2], h1[3]);
            bw[6] = hiHalf ? 0u : pk16(h1[4], h1[5]);
            bw[7] = hiHalf ? 0u : pk16(h1[6], h1[7]);
            v16h Bm = __builtin_bit_cast(v16h, bw);

            // 6 independent WMMAs: full pre-activations (biases via K=25;
            // r,z tiles pre-scaled by 0.5).
            v8f D0, D1, D2, D3, D4, D5;
            {
                v8f c = {};
                D0 = __builtin_amdgcn_wmma_f32_16x16x32_f16(false, A[0], false, Bm, (short)0, c, false, false);
                D1 = __builtin_amdgcn_wmma_f32_16x16x32_f16(false, A[1], false, Bm, (short)0, c, false, false);
                D2 = __builtin_amdgcn_wmma_f32_16x16x32_f16(false, A[2], false, Bm, (short)0, c, false, false);
                D3 = __builtin_amdgcn_wmma_f32_16x16x32_f16(false, A[3], false, Bm, (short)0, c, false, false);
                D4 = __builtin_amdgcn_wmma_f32_16x16x32_f16(false, A[4], false, Bm, (short)0, c, false, false);
                D5 = __builtin_amdgcn_wmma_f32_16x16x32_f16(false, A[5], false, Bm, (short)0, c, false, false);
            }

            // Gate math + state update, per-lane (positions align across tiles).
            #pragma unroll
            for (int j = 0; j < 8; ++j) {
                float r0 = sigmoid_half_pre(D0[j]);    // tanh + fma
                float z0 = sigmoid_half_pre(D2[j]);
                float n0 = fast_tanh(fmaf(xv, wn0[j], bn0[j]) + r0 * D4[j]);
                h0[j] = fmaf(z0, h0[j] - n0, n0);      // (1-z)*n + z*h

                float r1 = sigmoid_half_pre(D1[j]);
                float z1 = sigmoid_half_pre(D3[j]);
                float n1 = fast_tanh(fmaf(xv, wn1[j], bn1[j]) + r1 * D5[j]);
                h1[j] = fmaf(z1, h1[j] - n1, n1);
            }
        }
    }

    // Final FC: out[b, o] = sum_u fc_w[o,u] * h[u,b] + fc_b[o]
    float acc0 = 0.0f, acc1 = 0.0f;
    #pragma unroll
    for (int j = 0; j < 8; ++j) {
        int u0 = hiHalf ? 16 + j : j;
        acc0 += fc_w[u0]      * h0[j];
        acc1 += fc_w[HH + u0] * h0[j];
        if (!hiHalf) {                         // h1 = units 8-15 (lane<16 only)
            acc0 += fc_w[8 + j]      * h1[j];
            acc1 += fc_w[HH + 8 + j] * h1[j];
        }
    }
    acc0 += __shfl_xor(acc0, 16, 32);
    acc1 += __shfl_xor(acc1, 16, 32);
    if (!hiHalf) {
        size_t b = (size_t)(rowbase + col);
        out[b * 2 + 0] = acc0 + fc_b[0];
        out[b * 2 + 1] = acc1 + fc_b[1];
    }
}

extern "C" void kernel_launch(void* const* d_in, const int* in_sizes, int n_in,
                              void* d_out, int out_size, void* d_ws, size_t ws_size,
                              hipStream_t stream) {
    const float* x    = (const float*)d_in[0];
    const float* W_ih = (const float*)d_in[1];
    const float* b_ih = (const float*)d_in[2];
    const float* W_hh = (const float*)d_in[3];
    const float* b_hh = (const float*)d_in[4];
    const float* fc_w = (const float*)d_in[5];
    const float* fc_b = (const float*)d_in[6];
    float* out = (float*)d_out;

    int Bsz = in_sizes[0] / T_STEPS;          // x holds B*T floats
    int wavesNeeded   = (Bsz + 15) / 16;      // one wave per 16 batch rows
    int wavesPerBlock = 8;                    // 256 threads
    int blocks = (wavesNeeded + wavesPerBlock - 1) / wavesPerBlock;

    trendgru_wmma_kernel<<<blocks, wavesPerBlock * 32, 0, stream>>>(
        x, W_ih, b_ih, W_hh, b_hh, fc_w, fc_b, out, Bsz);
}
